// CausalSelfAttention_37417755083187
// MI455X (gfx1250) — matrix operand order repaired and checked by reading, not compiled, not run on hardware
//
#include <hip/hip_runtime.h>

// ---------------------------------------------------------------------------
// CDNA5 (gfx1250, wave32) causal self-attention forward:
//   q,k,v = x@W*, gated v, RoPE+RMSnorm, flash attention, out = y@Wproj
// Matrix work: v_wmma_f32_16x16x32_bf16. Staging: clause-friendly 128-bit
// global loads (all issued before converts), packed bf16 LDS stores,
// double-buffered LDS, global_prefetch.
// ---------------------------------------------------------------------------

typedef __attribute__((ext_vector_type(16))) __bf16 v16bf;
typedef __attribute__((ext_vector_type(4)))  __bf16 v4bf;
typedef __attribute__((ext_vector_type(8)))  float  v8f;

#define BB  2
#define TT  2048
#define CC  2048
#define NHH 16
#define HDD 128

static __device__ __forceinline__ __bf16 f2bf(float f) { return (__bf16)f; }

// A-fragment K index for 16-bit 16x32 A matrix (wave32):
// lanes 0-15 (hi=0): j0..7 -> K0..7,  j8..15 -> K16..23
// lanes16-31 (hi=1): j0..7 -> K8..15, j8..15 -> K24..31
static __device__ __forceinline__ int kmapA(int j, int hi) {
    return (j < 8) ? (hi * 8 + j) : (16 + hi * 8 + (j - 8));
}

// ---------------------------------------------------------------------------
// bf16 WMMA GEMM: C[M,N] = A[M,K] @ B[K,N]  (A,B,C f32 in memory)
// Block tile 64(M) x 128(N), 8 waves in 2(M) x 4(N) grid, 32x32 per wave:
// per K-step each wave: 2 A-frags + 2 B-frags -> 4 v_wmma.
// Double-buffered LDS; two-phase staging (loads first, then converts).
// ---------------------------------------------------------------------------
__global__ __launch_bounds__(256) void gemm_bf16_wmma(
    const float* __restrict__ A, const float* __restrict__ Bm,
    float* __restrict__ Cc, int M, int N, int K)
{
    __shared__ __align__(16) __bf16 As[2][64][32];    // [buf][m][k]   8 KB
    __shared__ __align__(16) __bf16 Bs[2][128][32];   // [buf][n][k]  16 KB

    const int tid  = threadIdx.x;
    const int wave = tid >> 5;
    const int lane = tid & 31;
    const int ln   = lane & 15;
    const int hi   = lane >> 4;
    const int mw   = wave & 1;    // 0..1 -> m rows mw*32..+31
    const int nw   = wave >> 1;   // 0..3 -> n cols nw*32..+31
    const int bm   = blockIdx.y * 64;
    const int bn   = blockIdx.x * 128;

    // per-thread staging coordinates (fixed across K-steps)
    const int ar0 = tid >> 3,          ac0 = (tid & 7) << 2;          // A elem 0
    const int ar1 = (tid + 256) >> 3,  ac1 = ac0;                     // A elem 1
    const int br  = tid >> 5;                                         // B k-row base
    const int bc  = (tid & 31) << 2;                                  // B n-col

    auto stage = [&](int bufi, int kk0) {
        // ---- phase 1: issue all global loads (claused, deep in flight) ----
        float4 fa0 = *reinterpret_cast<const float4*>(&A[(size_t)(bm + ar0) * K + kk0 + ac0]);
        float4 fa1 = *reinterpret_cast<const float4*>(&A[(size_t)(bm + ar1) * K + kk0 + ac1]);
        float4 fb[4];
        #pragma unroll
        for (int i = 0; i < 4; ++i)
            fb[i] = *reinterpret_cast<const float4*>(
                &Bm[(size_t)(kk0 + br + i * 8) * N + bn + bc]);
        if (kk0 + 64 < K) {
            __builtin_prefetch(&A[(size_t)(bm + ar0) * K + kk0 + 64 + ac0], 0, 1);
            __builtin_prefetch(&Bm[(size_t)(kk0 + 64 + br) * N + bn + bc], 0, 1);
        }
        // ---- phase 2: convert + LDS stores ----
        v4bf p0, p1;
        p0[0] = f2bf(fa0.x); p0[1] = f2bf(fa0.y); p0[2] = f2bf(fa0.z); p0[3] = f2bf(fa0.w);
        p1[0] = f2bf(fa1.x); p1[1] = f2bf(fa1.y); p1[2] = f2bf(fa1.z); p1[3] = f2bf(fa1.w);
        *reinterpret_cast<v4bf*>(&As[bufi][ar0][ac0]) = p0;
        *reinterpret_cast<v4bf*>(&As[bufi][ar1][ac1]) = p1;
        #pragma unroll
        for (int i = 0; i < 4; ++i) {
            int r = br + i * 8;
            Bs[bufi][bc + 0][r] = f2bf(fb[i].x);
            Bs[bufi][bc + 1][r] = f2bf(fb[i].y);
            Bs[bufi][bc + 2][r] = f2bf(fb[i].z);
            Bs[bufi][bc + 3][r] = f2bf(fb[i].w);
        }
    };

    v8f acc[2][2];
    #pragma unroll
    for (int i = 0; i < 2; ++i)
        #pragma unroll
        for (int j = 0; j < 2; ++j)
            #pragma unroll
            for (int r = 0; r < 8; ++r) acc[i][j][r] = 0.0f;

    stage(0, 0);
    int buf = 0;
    for (int k0 = 0; k0 < K; k0 += 32) {
        __syncthreads();                       // staged tile visible / prev compute done
        if (k0 + 32 < K) stage(buf ^ 1, k0 + 32);

        v16bf a0, a1, b0, b1;
        #pragma unroll
        for (int j = 0; j < 16; ++j) {
            int kk = kmapA(j, hi);
            a0[j] = As[buf][mw * 32 + ln][kk];
            a1[j] = As[buf][mw * 32 + 16 + ln][kk];
        }
        #pragma unroll
        for (int j = 0; j < 16; ++j) {
            int kk = kmapA(j, hi);
            b0[j] = Bs[buf][nw * 32 + ln][kk];
            b1[j] = Bs[buf][nw * 32 + 16 + ln][kk];
        }
        acc[0][0] = __builtin_amdgcn_wmma_f32_16x16x32_bf16(
            false, a0, false, b0, (short)0, acc[0][0], false, false);
        acc[0][1] = __builtin_amdgcn_wmma_f32_16x16x32_bf16(
            false, a0, false, b1, (short)0, acc[0][1], false, false);
        acc[1][0] = __builtin_amdgcn_wmma_f32_16x16x32_bf16(
            false, a1, false, b0, (short)0, acc[1][0], false, false);
        acc[1][1] = __builtin_amdgcn_wmma_f32_16x16x32_bf16(
            false, a1, false, b1, (short)0, acc[1][1], false, false);
        buf ^= 1;
    }

    #pragma unroll
    for (int i = 0; i < 2; ++i)
        #pragma unroll
        for (int j = 0; j < 2; ++j)
            #pragma unroll
            for (int r = 0; r < 8; ++r) {
                int row = bm + mw * 32 + i * 16 + r + hi * 8;
                int col = bn + nw * 32 + j * 16 + ln;
                Cc[(size_t)row * N + col] = acc[i][j][r];
            }
}

// ---------------------------------------------------------------------------
// Fused post-QKV elementwise: gate = 2*sigmoid(x[:, :32]@Wgate); v += gate*ve;
// q,k <- rmsnorm(rope(q,k)).  One block (128 thr) per token.
// ---------------------------------------------------------------------------
__global__ __launch_bounds__(128) void qkv_post(
    const float* __restrict__ x, const float* __restrict__ ve,
    const float* __restrict__ cosb, const float* __restrict__ sinb,
    const float* __restrict__ Wgate,
    float* __restrict__ q, float* __restrict__ k, float* __restrict__ v)
{
    const int tid = threadIdx.x;
    const int bt  = blockIdx.x;          // b*T + t
    const int t   = bt & (TT - 1);
    const size_t base = (size_t)bt * CC;

    __shared__ float sgate[NHH];
    __shared__ float red[64];

    if (tid < NHH) {
        float g = 0.0f;
        #pragma unroll
        for (int gg = 0; gg < 32; ++gg)
            g += x[base + gg] * Wgate[gg * NHH + tid];
        sgate[tid] = 2.0f / (1.0f + __expf(-g));
    }
    __syncthreads();

    #pragma unroll
    for (int hh = 0; hh < NHH; ++hh) {
        size_t i = base + hh * HDD + tid;
        v[i] += sgate[hh] * ve[i];
    }

    for (int m = 0; m < 2; ++m) {
        float* ptr = m ? k : q;
        for (int hh = 0; hh < NHH; ++hh) {
            size_t hb = base + (size_t)hh * HDD;
            float r1 = 0.0f, r2 = 0.0f;
            if (tid < 64) {
                float x1 = ptr[hb + tid];
                float x2 = ptr[hb + 64 + tid];
                float cs = cosb[t * 64 + tid];
                float sn = sinb[t * 64 + tid];
                r1 = x1 * cs + x2 * sn;
                r2 = -x1 * sn + x2 * cs;
                red[tid] = r1 * r1 + r2 * r2;
            }
            __syncthreads();
            for (int ofs = 32; ofs >= 1; ofs >>= 1) {
                if (tid < ofs) red[tid] += red[tid + ofs];
                __syncthreads();
            }
            float rms = rsqrtf(red[0] / (float)HDD + 1e-6f);
            __syncthreads();
            if (tid < 64) {
                ptr[hb + tid]      = r1 * rms;
                ptr[hb + 64 + tid] = r2 * rms;
            }
            __syncthreads();
        }
    }
}

// ---------------------------------------------------------------------------
// Flash attention: block = 4 waves = (b, h, 64 q-rows); wave owns 16 q-rows.
// Stream 32-key chunks through LDS; online softmax; WMMA for qK^T and pV.
// ---------------------------------------------------------------------------
__global__ __launch_bounds__(128) void flash_attn(
    const float* __restrict__ q, const float* __restrict__ k,
    const float* __restrict__ v, float* __restrict__ y)
{
    __shared__ __align__(16) __bf16 Kb[32][HDD];      // [key][hd]
    __shared__ __align__(16) __bf16 Vt[HDD][32];      // [hd][key]
    __shared__ __align__(16) __bf16 Plds[4][16][32];  // per-wave P tile

    const int b    = blockIdx.z;
    const int h    = blockIdx.y;
    const int q0   = blockIdx.x * 64;
    const int tid  = threadIdx.x;
    const int wave = tid >> 5;
    const int lane = tid & 31;
    const int ln   = lane & 15;
    const int hi   = lane >> 4;
    const int qrow0 = q0 + wave * 16;

    const float scale = 0.08838834764831845f;   // 1/sqrt(128)

    // persistent scaled Q A-fragments (HD=128 -> 4 chunks of K=32)
    v16bf qf[4];
    const size_t qrowbase = ((size_t)(b * TT) + qrow0 + ln) * CC + (size_t)h * HDD;
    #pragma unroll
    for (int c = 0; c < 4; ++c)
        #pragma unroll
        for (int j = 0; j < 16; ++j)
            qf[c][j] = f2bf(q[qrowbase + c * 32 + kmapA(j, hi)] * scale);

    float mrow[8], lrow[8];
    v8f acc[8];
    #pragma unroll
    for (int r = 0; r < 8; ++r) { mrow[r] = -3.0e38f; lrow[r] = 0.0f; }
    #pragma unroll
    for (int cc = 0; cc < 8; ++cc)
        #pragma unroll
        for (int r = 0; r < 8; ++r) acc[cc][r] = 0.0f;

    const int nChunks = (q0 >> 5) + 2;   // causal: chunks with kbase <= q0+63
    for (int kc = 0; kc < nChunks; ++kc) {
        const int kbase = kc * 32;

        // cooperative stage of K (row-major, packed stores) and V (transposed),
        // two half-passes; each issues 8 global_load_b128 before any convert.
        #pragma unroll
        for (int half = 0; half < 2; ++half) {
            float4 fk[4], fv[4];
            #pragma unroll
            for (int i = 0; i < 4; ++i) {
                int e  = tid + (half * 4 + i) * 128;
                int r  = e >> 5;          // key 0..31
                int c4 = (e & 31) << 2;   // hd
                size_t g = ((size_t)(b * TT) + kbase + r) * CC + (size_t)h * HDD + c4;
                fk[i] = *reinterpret_cast<const float4*>(&k[g]);
                fv[i] = *reinterpret_cast<const float4*>(&v[g]);
            }
            #pragma unroll
            for (int i = 0; i < 4; ++i) {
                int e  = tid + (half * 4 + i) * 128;
                int r  = e >> 5;
                int c4 = (e & 31) << 2;
                v4bf pk;
                pk[0] = f2bf(fk[i].x); pk[1] = f2bf(fk[i].y);
                pk[2] = f2bf(fk[i].z); pk[3] = f2bf(fk[i].w);
                *reinterpret_cast<v4bf*>(&Kb[r][c4]) = pk;
                Vt[c4 + 0][r] = f2bf(fv[i].x);
                Vt[c4 + 1][r] = f2bf(fv[i].y);
                Vt[c4 + 2][r] = f2bf(fv[i].z);
                Vt[c4 + 3][r] = f2bf(fv[i].w);
            }
        }
        __syncthreads();

        // S = (q*scale) @ K^T : two 16x16 tiles covering 32 keys
        v8f s0, s1;
        #pragma unroll
        for (int r = 0; r < 8; ++r) { s0[r] = 0.0f; s1[r] = 0.0f; }
        #pragma unroll
        for (int c = 0; c < 4; ++c) {
            v16bf kb0, kb1;
            #pragma unroll
            for (int j = 0; j < 16; ++j) {
                int hd = c * 32 + kmapA(j, hi);
                kb0[j] = Kb[ln][hd];
                kb1[j] = Kb[16 + ln][hd];
            }
            s0 = __builtin_amdgcn_wmma_f32_16x16x32_bf16(
                false, qf[c], false, kb0, (short)0, s0, false, false);
            s1 = __builtin_amdgcn_wmma_f32_16x16x32_bf16(
                false, qf[c], false, kb1, (short)0, s1, false, false);
        }

        // online softmax (row spread across 16 lanes in D layout)
        const int key0 = kbase + ln;
        const int key1 = kbase + 16 + ln;
        float esc[8];
        #pragma unroll
        for (int r = 0; r < 8; ++r) {
            const int qrow = qrow0 + r + hi * 8;
            const bool ok0 = key0 <= qrow;
            const bool ok1 = key1 <= qrow;
            float v0 = ok0 ? s0[r] : -3.0e38f;
            float v1 = ok1 ? s1[r] : -3.0e38f;
            float mx = fmaxf(v0, v1);
            #pragma unroll
            for (int o = 1; o < 16; o <<= 1)
                mx = fmaxf(mx, __shfl_xor(mx, o, 16));
            float mnew = fmaxf(mrow[r], mx);
            esc[r] = __expf(mrow[r] - mnew);
            float e0 = ok0 ? __expf(v0 - mnew) : 0.0f;
            float e1 = ok1 ? __expf(v1 - mnew) : 0.0f;
            float rs = e0 + e1;
            #pragma unroll
            for (int o = 1; o < 16; o <<= 1)
                rs += __shfl_xor(rs, o, 16);
            lrow[r] = lrow[r] * esc[r] + rs;
            mrow[r] = mnew;
            Plds[wave][r + hi * 8][ln]      = f2bf(e0);
            Plds[wave][r + hi * 8][16 + ln] = f2bf(e1);
        }
        #pragma unroll
        for (int cc = 0; cc < 8; ++cc)
            #pragma unroll
            for (int r = 0; r < 8; ++r) acc[cc][r] *= esc[r];

        // reload P as A-fragment (re-layout through per-wave LDS slab)
        v16bf pf;
        #pragma unroll
        for (int j = 0; j < 16; ++j)
            pf[j] = Plds[wave][ln][kmapA(j, hi)];

        // acc += P @ V  (8 hd-chunks of 16 columns)
        #pragma unroll
        for (int cc = 0; cc < 8; ++cc) {
            v16bf vf;
            #pragma unroll
            for (int j = 0; j < 16; ++j)
                vf[j] = Vt[cc * 16 + ln][kmapA(j, hi)];
            acc[cc] = __builtin_amdgcn_wmma_f32_16x16x32_bf16(
                false, pf, false, vf, (short)0, acc[cc], false, false);
        }
        __syncthreads();
    }

    // epilogue: y[b, q, h*HD + hd] = acc / l
    #pragma unroll
    for (int cc = 0; cc < 8; ++cc)
        #pragma unroll
        for (int r = 0; r < 8; ++r) {
            int qrow = qrow0 + r + hi * 8;
            y[((size_t)(b * TT) + qrow) * CC + (size_t)h * HDD + cc * 16 + ln] =
                acc[cc][r] / lrow[r];
        }
}

// ---------------------------------------------------------------------------
extern "C" void kernel_launch(void* const* d_in, const int* in_sizes, int n_in,
                              void* d_out, int out_size, void* d_ws, size_t ws_size,
                              hipStream_t stream)
{
    (void)in_sizes; (void)n_in; (void)out_size; (void)ws_size;
    const float* x     = (const float*)d_in[0];
    const float* ve    = (const float*)d_in[1];
    const float* cosb  = (const float*)d_in[2];
    const float* sinb  = (const float*)d_in[3];
    const float* Wq    = (const float*)d_in[4];
    const float* Wk    = (const float*)d_in[5];
    const float* Wv    = (const float*)d_in[6];
    const float* Wgate = (const float*)d_in[7];
    const float* Wproj = (const float*)d_in[8];
    float* out = (float*)d_out;

    const int M = BB * TT;   // 4096
    const int N = CC;        // 2048
    const int K = CC;        // 2048

    float* qb = (float*)d_ws;
    float* kb = qb + (size_t)M * N;
    float* vb = kb + (size_t)M * N;
    float* yb = vb + (size_t)M * N;

    dim3 gg(N / 128, M / 64);   // 16 x 64
    gemm_bf16_wmma<<<gg, 256, 0, stream>>>(x, Wq, qb, M, N, K);
    gemm_bf16_wmma<<<gg, 256, 0, stream>>>(x, Wk, kb, M, N, K);
    gemm_bf16_wmma<<<gg, 256, 0, stream>>>(x, Wv, vb, M, N, K);

    qkv_post<<<M, 128, 0, stream>>>(x, ve, cosb, sinb, Wgate, qb, kb, vb);

    flash_attn<<<dim3(TT / 64, NHH, BB), 128, 0, stream>>>(qb, kb, vb, yb);

    gemm_bf16_wmma<<<gg, 256, 0, stream>>>(yb, Wproj, out, M, N, K);
}
